// TransformerBlock_69870527972004
// MI455X (gfx1250) — matrix operand order repaired and checked by reading, not compiled, not run on hardware
//
#include <hip/hip_runtime.h>

typedef __bf16 bf16_t;
typedef __attribute__((ext_vector_type(16))) __bf16 v16bf;
typedef __attribute__((ext_vector_type(8)))  __bf16 v8bf;
typedef __attribute__((ext_vector_type(8)))  float  v8f;
typedef __attribute__((ext_vector_type(4)))  unsigned int v4u;
typedef __attribute__((ext_vector_type(8)))  int v8i;
typedef __attribute__((ext_vector_type(4)))  int v4i;

#define D_MODEL 2048
#define HEAD_DIM 128
#define LN_EPS 1e-5f

static __device__ __forceinline__ v16bf cat8(v8bf lo, v8bf hi) {
  return __builtin_shufflevector(lo, hi, 0,1,2,3,4,5,6,7,8,9,10,11,12,13,14,15);
}
static __device__ __forceinline__ v8f wmma_bf16(v16bf a, v16bf b, v8f c) {
  // D = A(16x32 bf16) * B(32x16 bf16) + C(16x16 f32)
  return __builtin_amdgcn_wmma_f32_16x16x32_bf16(false, a, false, b, (short)0, c, false, false);
}
static __device__ __forceinline__ unsigned lds_off(const void* p) {
  // LDS aperture: flat addr[31:0] carries the LDS byte offset (ISA 10.2)
  return (unsigned)(unsigned long long)p;
}

// ---------------------------------------------------------------------------
// TDM: DMA one 2D bf16 tile (tile_d0 x tile_d1) from global to LDS, with
// row padding so that LDS row stride = tile_d0*2B + pad DWORDs*4B.
//   pad_iv: pad_interval field (2^(v+1) DWORDs of data before each pad)
//   pad_am: pad_amount field  (field+1 DWORDs of padding)
// Descriptor packing per CDNA5 ISA.
// Toolchain uses the 6-arg builtin: (v4u, v8i, v4i, v4i, v8i, i32 cpol).
// ---------------------------------------------------------------------------
static __device__ __forceinline__ void tdm_load_2d(
    const void* gptr, unsigned lds_addr,
    unsigned tensor_d0, unsigned tensor_d1,
    unsigned tile_d0, unsigned tile_d1,
    unsigned stride_d0, unsigned pad_iv, unsigned pad_am) {
  unsigned long long ga = (unsigned long long)gptr;
  v4u g0;
  g0[0] = 1u;                                    // count=1, user descriptor
  g0[1] = lds_addr;                              // LDS byte address
  g0[2] = (unsigned)ga;                          // global_addr[31:0]
  g0[3] = (unsigned)(ga >> 32) | (2u << 30);     // global_addr[56:32] | type=2
  v8i g1;
  g1[0] = (int)((1u << 16) |                     // data_size = 1 (2 bytes)
                (1u << 20) |                     // pad_enable
                (pad_iv << 22) | (pad_am << 25));
  g1[1] = (int)((tensor_d0 & 0xffffu) << 16);    // tensor_dim0[15:0]
  g1[2] = (int)((tensor_d0 >> 16) | ((tensor_d1 & 0xffffu) << 16));
  g1[3] = (int)((tensor_d1 >> 16) | (tile_d0 << 16));
  g1[4] = (int)tile_d1;                          // tile_dim1 | tile_dim2=0
  g1[5] = (int)stride_d0;                        // tensor_dim0_stride[31:0]
  g1[6] = 0;
  g1[7] = 0;
  v4i z4 = {0, 0, 0, 0};
  v8i z8 = {0, 0, 0, 0, 0, 0, 0, 0};
  __builtin_amdgcn_tensor_load_to_lds(g0, g1, z4, z4, z8, 0);
}

// ---------------------------------------------------------------------------
// fp32 -> bf16 elementwise convert (weights)
// ---------------------------------------------------------------------------
__global__ void cvt_bf16_kernel(const float* __restrict__ in, bf16_t* __restrict__ out, int n) {
  int i = blockIdx.x * 256 + threadIdx.x;
  if (i < n) out[i] = (bf16_t)in[i];
}

// ---------------------------------------------------------------------------
// LayerNorm over D_MODEL, output bf16
// ---------------------------------------------------------------------------
__global__ __launch_bounds__(256) void layernorm_bf16_kernel(
    const float* __restrict__ x, const float* __restrict__ w,
    const float* __restrict__ b, bf16_t* __restrict__ out) {
  __shared__ float red[256];
  const int row = blockIdx.x;
  const float* xr = x + (long)row * D_MODEL;
  float s = 0.f, s2 = 0.f;
  for (int i = threadIdx.x; i < D_MODEL; i += 256) { float v = xr[i]; s += v; s2 += v * v; }
  red[threadIdx.x] = s; __syncthreads();
  for (int o = 128; o > 0; o >>= 1) { if (threadIdx.x < o) red[threadIdx.x] += red[threadIdx.x + o]; __syncthreads(); }
  const float mu = red[0] * (1.0f / D_MODEL);
  __syncthreads();
  red[threadIdx.x] = s2; __syncthreads();
  for (int o = 128; o > 0; o >>= 1) { if (threadIdx.x < o) red[threadIdx.x] += red[threadIdx.x + o]; __syncthreads(); }
  const float var = red[0] * (1.0f / D_MODEL) - mu * mu;
  const float rs = rsqrtf(var + LN_EPS);
  bf16_t* orow = out + (long)row * D_MODEL;
  for (int i = threadIdx.x; i < D_MODEL; i += 256)
    orow[i] = (bf16_t)((xr[i] - mu) * rs * w[i] + b[i]);
}

// ---------------------------------------------------------------------------
// C = act(scale*(A @ W^T + bias)) [+ resid]; A:[M,K] bf16, W:[N,K] bf16
// block tile 128x128, BK=32, 256 threads = 8 waves (2x4), wave tile 64x32.
// Tiles are fed by the Tensor Data Mover (double-buffered, TENSORcnt).
// ---------------------------------------------------------------------------
template<bool OUT_BF16, bool RELU, bool RES, bool SCALE>
__global__ __launch_bounds__(256) void gemm_wmma_kernel(
    const bf16_t* __restrict__ A, const bf16_t* __restrict__ W,
    const float* __restrict__ bias, const float* __restrict__ resid,
    void* __restrict__ out, int M, int N, int K, float scale) {
  __shared__ bf16_t As[2][128][40];   // TDM pads 16DW rows by 4DW -> stride 40
  __shared__ bf16_t Ws[2][128][40];
  const int tid = threadIdx.x;
  const int lane = tid & 31, wave = tid >> 5;
  const int laneLo = lane & 15, laneHi = lane >> 4;
  const int wr = wave >> 2, wc = wave & 3;           // 2 x 4 wave grid
  const long blockM = (long)blockIdx.y * 128, blockN = (long)blockIdx.x * 128;
  const bf16_t* Ablk = A + blockM * (long)K;
  const bf16_t* Wblk = W + blockN * (long)K;

  v8f acc[4][2] = {};

  if (tid == 0) {      // prologue: DMA first K-tile (TDM ignores EXEC)
    tdm_load_2d(Ablk, lds_off(&As[0][0][0]), K, M, 32, 128, K, 3, 3);
    tdm_load_2d(Wblk, lds_off(&Ws[0][0][0]), K, N, 32, 128, K, 3, 3);
    __builtin_amdgcn_s_wait_tensorcnt(0);
  }
  __syncthreads();

  int cur = 0;
  for (int k0 = 0; k0 < K; k0 += 32) {
    if (tid == 0 && k0 + 32 < K) {    // prefetch next tile into other buffer
      tdm_load_2d(Ablk + k0 + 32, lds_off(&As[cur ^ 1][0][0]), K, M, 32, 128, K, 3, 3);
      tdm_load_2d(Wblk + k0 + 32, lds_off(&Ws[cur ^ 1][0][0]), K, N, 32, 128, K, 3, 3);
    }
    v16bf af[4], bfv[2];
#pragma unroll
    for (int mt = 0; mt < 4; ++mt) {
      int r = wr * 64 + mt * 16 + laneLo;
      af[mt] = cat8(*(const v8bf*)&As[cur][r][laneHi * 8],
                    *(const v8bf*)&As[cur][r][16 + laneHi * 8]);
    }
#pragma unroll
    for (int nt = 0; nt < 2; ++nt) {
      int n = wc * 32 + nt * 16 + laneLo;
      bfv[nt] = cat8(*(const v8bf*)&Ws[cur][n][laneHi * 8],
                     *(const v8bf*)&Ws[cur][n][16 + laneHi * 8]);
    }
#pragma unroll
    for (int mt = 0; mt < 4; ++mt)
#pragma unroll
      for (int nt = 0; nt < 2; ++nt)
        acc[mt][nt] = wmma_bf16(af[mt], bfv[nt], acc[mt][nt]);
    if (tid == 0) __builtin_amdgcn_s_wait_tensorcnt(0);
    __syncthreads();
    cur ^= 1;
  }
#pragma unroll
  for (int mt = 0; mt < 4; ++mt)
#pragma unroll
    for (int nt = 0; nt < 2; ++nt)
#pragma unroll
      for (int r = 0; r < 8; ++r) {
        long row = blockM + wr * 64 + mt * 16 + laneHi * 8 + r;
        long col = blockN + wc * 32 + nt * 16 + laneLo;
        float v = acc[mt][nt][r] + bias[col];
        if (SCALE) v *= scale;
        if (RELU)  v = fmaxf(v, 0.f);
        long idx = row * N + col;
        if (RES) v += resid[idx];
        if (OUT_BF16) ((bf16_t*)out)[idx] = (bf16_t)v;
        else          ((float*)out)[idx] = v;
      }
}

// ---------------------------------------------------------------------------
// Flash attention: one block = (b, h, 64 query rows), loop 64-key tiles.
// Q/K tiles fed by TDM; V transposed into LDS by the threads.
// q is already pre-scaled by HD^-0.5 in the Q projection epilogue.
// ---------------------------------------------------------------------------
__global__ __launch_bounds__(256) void flash_attn_kernel(
    const bf16_t* __restrict__ q, const bf16_t* __restrict__ k,
    const bf16_t* __restrict__ v, bf16_t* __restrict__ o, int S) {
  __shared__ bf16_t Qs[64][136];   // TDM pads 64DW rows by 4DW -> stride 136
  __shared__ bf16_t Ks[64][136];
  __shared__ bf16_t Vs[128][72];   // transposed: Vs[d][kk], 144B stride
  __shared__ float  Ss[64][68];
  __shared__ bf16_t Ps[64][72];
  __shared__ float  m_s[64], l_s[64], alpha_s[64];

  const int tid = threadIdx.x;
  const int lane = tid & 31, wave = tid >> 5;
  const int laneLo = lane & 15, laneHi = lane >> 4;
  const int q0 = blockIdx.x * 64;
  const long base = ((long)blockIdx.z * S) * D_MODEL + blockIdx.y * HEAD_DIM;

  if (tid == 0)   // Q tile: 64 rows x 128 bf16, row stride D_MODEL
    tdm_load_2d(q + base + (long)q0 * D_MODEL, lds_off(&Qs[0][0]),
                D_MODEL, 64, 128, 64, D_MODEL, 5, 3);
  if (tid < 64) { m_s[tid] = -3.0e38f; l_s[tid] = 0.f; }

  const int wrO = wave >> 2, wcO = wave & 3;          // 2x4 over 64x128 output
  v8f oacc[2][2] = {};

  for (int kt = 0; kt < S; kt += 64) {
    __syncthreads();                                  // prior reads of Ks/Vs done
    if (tid == 0)
      tdm_load_2d(k + base + (long)kt * D_MODEL, lds_off(&Ks[0][0]),
                  D_MODEL, 64, 128, 64, D_MODEL, 5, 3);
#pragma unroll
    for (int i = 0; i < 4; ++i) {                     // V tile, transposed
      int c = tid + i * 256;
      int r = c >> 4, d0 = (c & 15) * 8;
      v8bf vv = *(const v8bf*)(v + base + (long)(kt + r) * D_MODEL + d0);
#pragma unroll
      for (int j = 0; j < 8; ++j) Vs[d0 + j][r] = vv[j];
    }
    if (tid == 0) __builtin_amdgcn_s_wait_tensorcnt(0);
    __syncthreads();

    // S = Q K^T : wave grid 2x4, wave tile 32 rows x 16 cols, K-dim = 128
    v8f sacc[2] = {};
#pragma unroll
    for (int ds = 0; ds < 4; ++ds) {
      int n = (wave & 3) * 16 + laneLo;
      v16bf bk = cat8(*(const v8bf*)&Ks[n][ds * 32 + laneHi * 8],
                      *(const v8bf*)&Ks[n][ds * 32 + 16 + laneHi * 8]);
#pragma unroll
      for (int mt = 0; mt < 2; ++mt) {
        int r = (wave >> 2) * 32 + mt * 16 + laneLo;
        v16bf aq = cat8(*(const v8bf*)&Qs[r][ds * 32 + laneHi * 8],
                        *(const v8bf*)&Qs[r][ds * 32 + 16 + laneHi * 8]);
        sacc[mt] = wmma_bf16(aq, bk, sacc[mt]);
      }
    }
#pragma unroll
    for (int mt = 0; mt < 2; ++mt)
#pragma unroll
      for (int r = 0; r < 8; ++r)
        Ss[(wave >> 2) * 32 + mt * 16 + laneHi * 8 + r][(wave & 3) * 16 + laneLo] = sacc[mt][r];
    __syncthreads();

    // online softmax: one thread per query row
    if (tid < 64) {
      float m_old = m_s[tid], rmax = m_old;
      for (int c = 0; c < 64; ++c) rmax = fmaxf(rmax, Ss[tid][c]);
      float al = __expf(m_old - rmax);
      float sum = 0.f;
      for (int c = 0; c < 64; ++c) {
        float p = __expf(Ss[tid][c] - rmax);
        Ps[tid][c] = (bf16_t)p;
        sum += p;
      }
      m_s[tid] = rmax;
      alpha_s[tid] = al;
      l_s[tid] = l_s[tid] * al + sum;
    }
    __syncthreads();

    // rescale running O, accumulate P @ V (wave tile 32x32, K-dim = 64)
#pragma unroll
    for (int mt = 0; mt < 2; ++mt)
#pragma unroll
      for (int nt = 0; nt < 2; ++nt)
#pragma unroll
        for (int r = 0; r < 8; ++r)
          oacc[mt][nt][r] *= alpha_s[wrO * 32 + mt * 16 + laneHi * 8 + r];
#pragma unroll
    for (int ks = 0; ks < 2; ++ks) {
      v16bf ap[2], bv[2];
#pragma unroll
      for (int mt = 0; mt < 2; ++mt) {
        int r = wrO * 32 + mt * 16 + laneLo;
        ap[mt] = cat8(*(const v8bf*)&Ps[r][ks * 32 + laneHi * 8],
                      *(const v8bf*)&Ps[r][ks * 32 + 16 + laneHi * 8]);
      }
#pragma unroll
      for (int nt = 0; nt < 2; ++nt) {
        int dcol = wcO * 32 + nt * 16 + laneLo;
        bv[nt] = cat8(*(const v8bf*)&Vs[dcol][ks * 32 + laneHi * 8],
                      *(const v8bf*)&Vs[dcol][ks * 32 + 16 + laneHi * 8]);
      }
#pragma unroll
      for (int mt = 0; mt < 2; ++mt)
#pragma unroll
        for (int nt = 0; nt < 2; ++nt)
          oacc[mt][nt] = wmma_bf16(ap[mt], bv[nt], oacc[mt][nt]);
    }
  }
  __syncthreads();
#pragma unroll
  for (int mt = 0; mt < 2; ++mt)
#pragma unroll
    for (int nt = 0; nt < 2; ++nt)
#pragma unroll
      for (int r = 0; r < 8; ++r) {
        int rl = wrO * 32 + mt * 16 + laneHi * 8 + r;
        int dcol = wcO * 32 + nt * 16 + laneLo;
        o[base + (long)(q0 + rl) * D_MODEL + dcol] = (bf16_t)(oacc[mt][nt][r] / l_s[rl]);
      }
}

// ---------------------------------------------------------------------------
extern "C" void kernel_launch(void* const* d_in, const int* in_sizes, int n_in,
                              void* d_out, int out_size, void* d_ws, size_t ws_size,
                              hipStream_t stream) {
  (void)in_sizes; (void)n_in; (void)out_size; (void)ws_size;
  const float* hidden = (const float*)d_in[0];
  const float* ln1w = (const float*)d_in[1];
  const float* ln1b = (const float*)d_in[2];
  const float* wq = (const float*)d_in[3];
  const float* bq = (const float*)d_in[4];
  const float* wk = (const float*)d_in[5];
  const float* bk = (const float*)d_in[6];
  const float* wv = (const float*)d_in[7];
  const float* bv = (const float*)d_in[8];
  const float* wo = (const float*)d_in[9];
  const float* bo = (const float*)d_in[10];
  const float* ln2w = (const float*)d_in[11];
  const float* ln2b = (const float*)d_in[12];
  const float* w1 = (const float*)d_in[13];
  const float* b1 = (const float*)d_in[14];
  const float* w2 = (const float*)d_in[15];
  const float* b2 = (const float*)d_in[16];

  const int S = 2048, Dm = D_MODEL, F = 8192;
  const int M = 2 * S;  // 4096 rows

  char* ws = (char*)d_ws;
  size_t off = 0;
  auto alloc = [&](size_t bytes) { char* p = ws + off; off += (bytes + 255) & ~(size_t)255; return p; };
  bf16_t* xln  = (bf16_t*)alloc((size_t)M * Dm * 2);
  bf16_t* wqb  = (bf16_t*)alloc((size_t)Dm * Dm * 2);
  bf16_t* wkb  = (bf16_t*)alloc((size_t)Dm * Dm * 2);
  bf16_t* wvb  = (bf16_t*)alloc((size_t)Dm * Dm * 2);
  bf16_t* wob  = (bf16_t*)alloc((size_t)Dm * Dm * 2);
  bf16_t* w1b  = (bf16_t*)alloc((size_t)F * Dm * 2);
  bf16_t* w2b  = (bf16_t*)alloc((size_t)Dm * F * 2);
  bf16_t* qb   = (bf16_t*)alloc((size_t)M * Dm * 2);
  bf16_t* kb   = (bf16_t*)alloc((size_t)M * Dm * 2);
  bf16_t* vb   = (bf16_t*)alloc((size_t)M * Dm * 2);
  bf16_t* attn = (bf16_t*)alloc((size_t)M * Dm * 2);
  float*  hbuf = (float*)alloc((size_t)M * Dm * 4);
  bf16_t* hln  = (bf16_t*)alloc((size_t)M * Dm * 2);
  bf16_t* mid  = (bf16_t*)alloc((size_t)M * F * 2);

  // weights -> bf16
  const int nDD = Dm * Dm, nFD = F * Dm;
  cvt_bf16_kernel<<<(nDD + 255) / 256, 256, 0, stream>>>(wq, wqb, nDD);
  cvt_bf16_kernel<<<(nDD + 255) / 256, 256, 0, stream>>>(wk, wkb, nDD);
  cvt_bf16_kernel<<<(nDD + 255) / 256, 256, 0, stream>>>(wv, wvb, nDD);
  cvt_bf16_kernel<<<(nDD + 255) / 256, 256, 0, stream>>>(wo, wob, nDD);
  cvt_bf16_kernel<<<(nFD + 255) / 256, 256, 0, stream>>>(w1, w1b, nFD);
  cvt_bf16_kernel<<<(nFD + 255) / 256, 256, 0, stream>>>(w2, w2b, nFD);

  // LN1
  layernorm_bf16_kernel<<<M, 256, 0, stream>>>(hidden, ln1w, ln1b, xln);

  // Q/K/V projections (Q pre-scaled by HD^-0.5)
  const float qscale = 0.08838834764831845f;  // 128^-0.5
  dim3 gDD(Dm / 128, M / 128);
  gemm_wmma_kernel<true, false, false, true ><<<gDD, 256, 0, stream>>>(xln, wqb, bq, nullptr, qb, M, Dm, Dm, qscale);
  gemm_wmma_kernel<true, false, false, false><<<gDD, 256, 0, stream>>>(xln, wkb, bk, nullptr, kb, M, Dm, Dm, 1.f);
  gemm_wmma_kernel<true, false, false, false><<<gDD, 256, 0, stream>>>(xln, wvb, bv, nullptr, vb, M, Dm, Dm, 1.f);

  // attention
  flash_attn_kernel<<<dim3(S / 64, 16, 2), 256, 0, stream>>>(qb, kb, vb, attn, S);

  // output projection + residual (fp32 h)
  gemm_wmma_kernel<false, false, true, false><<<gDD, 256, 0, stream>>>(attn, wob, bo, hidden, hbuf, M, Dm, Dm, 1.f);

  // LN2
  layernorm_bf16_kernel<<<M, 256, 0, stream>>>(hbuf, ln2w, ln2b, hln);

  // FFN
  gemm_wmma_kernel<true, true, false, false><<<dim3(F / 128, M / 128), 256, 0, stream>>>(hln, w1b, b1, nullptr, mid, M, F, Dm, 1.f);
  gemm_wmma_kernel<false, false, true, false><<<gDD, 256, 0, stream>>>(mid, w2b, b2, hbuf, (float*)d_out, M, Dm, F, 1.f);
}
